// MultiHeadAttentionNN_50508815401447
// MI455X (gfx1250) — matrix operand order repaired and checked by reading, not compiled, not run on hardware
//
#include <hip/hip_runtime.h>

// ---------------------------------------------------------------------------
// Multi-head attention forward for MI455X (gfx1250, wave32, WMMA).
//   Stage 0: fp32 -> bf16 conversion of x, qkv_w, out_w (native v_cvt).
//   Stage 1: QKV GEMM   (4096x3072x1024)  bf16 WMMA, bf16 out.
//   Stage 2: Flash attention, 128-query blocks, double-buffered 32-key LDS
//            tiles with pipelined async global->LDS K staging (ASYNCcnt),
//            DPP16 softmax reductions.
//   Stage 3: Output GEMM (4096x1024x1024) bf16 WMMA, f32 out + bias.
// ---------------------------------------------------------------------------

typedef __attribute__((ext_vector_type(16))) __bf16 v16bf;
typedef __attribute__((ext_vector_type(8)))  float  v8f;

union BF16Frag {
  v16bf v;
  uint4 q[2];
  unsigned short s[16];
};

// Native hardware convert (v_cvt_pk_bf16_f32 class), RNE.
__device__ __forceinline__ unsigned short f2bf(float f) {
  union { __bf16 h; unsigned short u; } c;
  c.h = (__bf16)f;
  return c.u;
}

__device__ __forceinline__ v8f wmma_bf16(v16bf a, v16bf b, v8f c) {
  return __builtin_amdgcn_wmma_f32_16x16x32_bf16(false, a, false, b, (short)0, c,
                                                 false, false);
}

// Butterfly reduction across each 16-lane half via DPP16 (no LDS, no waits).
// xor-type permutes {1, 2, 7, 15} span the 4-bit lane space.
__device__ __forceinline__ float dpp_max16(float x) {
  union { float f; int i; } a, t;
  a.f = x;
  t.i = __builtin_amdgcn_update_dpp(0, a.i, 0x0B1, 0xF, 0xF, true); // quad_perm [1,0,3,2]
  a.f = fmaxf(a.f, t.f);
  t.i = __builtin_amdgcn_update_dpp(0, a.i, 0x04E, 0xF, 0xF, true); // quad_perm [2,3,0,1]
  a.f = fmaxf(a.f, t.f);
  t.i = __builtin_amdgcn_update_dpp(0, a.i, 0x141, 0xF, 0xF, true); // row_half_mirror
  a.f = fmaxf(a.f, t.f);
  t.i = __builtin_amdgcn_update_dpp(0, a.i, 0x140, 0xF, 0xF, true); // row_mirror
  a.f = fmaxf(a.f, t.f);
  return a.f;
}

__device__ __forceinline__ float dpp_sum16(float x) {
  union { float f; int i; } a, t;
  a.f = x;
  t.i = __builtin_amdgcn_update_dpp(0, a.i, 0x0B1, 0xF, 0xF, true);
  a.f += t.f;
  t.i = __builtin_amdgcn_update_dpp(0, a.i, 0x04E, 0xF, 0xF, true);
  a.f += t.f;
  t.i = __builtin_amdgcn_update_dpp(0, a.i, 0x141, 0xF, 0xF, true);
  a.f += t.f;
  t.i = __builtin_amdgcn_update_dpp(0, a.i, 0x140, 0xF, 0xF, true);
  a.f += t.f;
  return a.f;
}

// ---------------------------------------------------------------------------
__global__ void __launch_bounds__(256)
cvt_f32_to_bf16(const float* __restrict__ src, unsigned short* __restrict__ dst,
                int n4) {
  int i = blockIdx.x * 256 + threadIdx.x;
  if (i >= n4) return;
  float4 f = ((const float4*)src)[i];
  uint2 o;
  o.x = (unsigned)f2bf(f.x) | ((unsigned)f2bf(f.y) << 16);
  o.y = (unsigned)f2bf(f.z) | ((unsigned)f2bf(f.w) << 16);
  ((uint2*)dst)[i] = o;
}

// ---------------------------------------------------------------------------
// C[M,N] = A[M,K] * W[N,K]^T  (row-major, K contiguous for both operands)
// 256 threads = 8 waves (2M x 4N); wave tile 32x64 = 2x4 WMMA accumulators.
// ---------------------------------------------------------------------------
template <bool OUT_F32>
__global__ void __launch_bounds__(256)
gemm_bf16_nt(const unsigned short* __restrict__ A,
             const unsigned short* __restrict__ W,
             void* __restrict__ outp,
             const float* __restrict__ bias,
             int M, int N, int K) {
  const int lane = threadIdx.x & 31;
  const int wid  = threadIdx.x >> 5;
  const int lc   = lane & 15;
  const int lh   = lane >> 4;
  const int wm   = wid & 1;
  const int wn   = wid >> 1;
  const int m0   = blockIdx.y * 64 + wm * 32;
  const int n0   = blockIdx.x * 256 + wn * 64;

  v8f acc[2][4];
#pragma unroll
  for (int i = 0; i < 2; ++i)
#pragma unroll
    for (int j = 0; j < 4; ++j) acc[i][j] = (v8f){0.f,0.f,0.f,0.f,0.f,0.f,0.f,0.f};

  const unsigned short* arow[2];
  arow[0] = A + (size_t)(m0 + lc) * K;
  arow[1] = A + (size_t)(m0 + 16 + lc) * K;
  const unsigned short* wrow[4];
#pragma unroll
  for (int j = 0; j < 4; ++j) wrow[j] = W + (size_t)(n0 + j * 16 + lc) * K;

  for (int k0 = 0; k0 < K; k0 += 32) {
    if (k0 + 256 < K) {
      __builtin_prefetch(arow[0] + k0 + 256, 0, 0);
      __builtin_prefetch(wrow[0] + k0 + 256, 0, 0);
    }
    BF16Frag a[2], b[4];
#pragma unroll
    for (int i = 0; i < 2; ++i) {
      a[i].q[0] = *(const uint4*)(arow[i] + k0 + lh * 16);
      a[i].q[1] = *(const uint4*)(arow[i] + k0 + lh * 16 + 8);
    }
#pragma unroll
    for (int j = 0; j < 4; ++j) {
      b[j].q[0] = *(const uint4*)(wrow[j] + k0 + lh * 16);
      b[j].q[1] = *(const uint4*)(wrow[j] + k0 + lh * 16 + 8);
    }
#pragma unroll
    for (int i = 0; i < 2; ++i)
#pragma unroll
      for (int j = 0; j < 4; ++j)
        acc[i][j] = wmma_bf16(a[i].v, b[j].v, acc[i][j]);
  }

#pragma unroll
  for (int i = 0; i < 2; ++i) {
    const int row0 = m0 + i * 16 + lh * 8;
#pragma unroll
    for (int j = 0; j < 4; ++j) {
      const int col = n0 + j * 16 + lc;
      if (OUT_F32) {
        float bs = bias[col];
        float* out = (float*)outp;
#pragma unroll
        for (int r = 0; r < 8; ++r)
          out[(size_t)(row0 + r) * N + col] = acc[i][j][r] + bs;
      } else {
        unsigned short* out = (unsigned short*)outp;
#pragma unroll
        for (int r = 0; r < 8; ++r)
          out[(size_t)(row0 + r) * N + col] = f2bf(acc[i][j][r]);
      }
    }
  }
}

// ---------------------------------------------------------------------------
// Flash attention. qkv: [B*S, 3*FEA] bf16; head h at cols [h*192, +192):
// q=0..63, k=64..127, v=128..191.  Block = 128 threads (4 waves) handles 128
// query rows of one (b,h); wave owns 32 rows (2 strips of 16).
// K/V tiles double-buffered in LDS; K staged by pipelined async global->LDS
// (issue tile kb+1 while computing tile kb, partial s_wait_asynccnt).
// ---------------------------------------------------------------------------
__global__ void __launch_bounds__(128)
attn_flash(const unsigned short* __restrict__ qkv,
           unsigned short* __restrict__ out) {
  constexpr int S = 2048, LDQ = 3072, FEA = 1024;
  constexpr int NKB = S / 32;                // 64 key tiles

  __shared__ unsigned short kT[2][32][80];   // K tiles  [key][d]  (160B rows)
  __shared__ unsigned short vT[2][64][48];   // V^T tiles [d][key] (96B rows)
  __shared__ unsigned short pB[4][32][32];   // per-wave P tile [q][key]

  const int lane = threadIdx.x & 31;
  const int wid  = threadIdx.x >> 5;
  const int lc   = lane & 15;
  const int lh   = lane >> 4;
  const int qblk = blockIdx.x;     // 0..15
  const int h    = blockIdx.y;     // 0..15
  const int b    = blockIdx.z;     // 0..1
  const int q0   = qblk * 128 + wid * 32;
  const size_t rowBase = (size_t)b * S;

  // Issue one K tile's async global->LDS copies (2 b128 ops per wave).
  auto issueK = [&](int kb) {
    const int key0 = kb * 32;
    unsigned short (*kbuf)[80] = kT[kb & 1];
    for (int i = threadIdx.x; i < 256; i += 128) {
      const int kr = i >> 3, seg = i & 7;
      const unsigned short* src =
          qkv + (rowBase + key0 + kr) * LDQ + h * 192 + 64 + seg * 8;
      unsigned ldsOff = (unsigned)(size_t)&kbuf[kr][seg * 8];
      asm volatile("global_load_async_to_lds_b128 %0, %1, off"
                   :: "v"(ldsOff), "v"(src)
                   : "memory");
    }
  };

  // Q fragments: 2 strips x 2 d-chunks (16x32 A-fragments over HD=64).
  BF16Frag aq[2][2];
#pragma unroll
  for (int i = 0; i < 2; ++i) {
    const unsigned short* qr =
        qkv + (rowBase + q0 + i * 16 + lc) * LDQ + h * 192;
#pragma unroll
    for (int c = 0; c < 2; ++c) {
      aq[i][c].q[0] = *(const uint4*)(qr + c * 32 + lh * 16);
      aq[i][c].q[1] = *(const uint4*)(qr + c * 32 + lh * 16 + 8);
    }
  }

  v8f accO[2][4];
#pragma unroll
  for (int i = 0; i < 2; ++i)
#pragma unroll
    for (int nt = 0; nt < 4; ++nt)
      accO[i][nt] = (v8f){0.f,0.f,0.f,0.f,0.f,0.f,0.f,0.f};
  float m[2][8], l[2][8];
#pragma unroll
  for (int i = 0; i < 2; ++i)
#pragma unroll
    for (int r = 0; r < 8; ++r) { m[i][r] = -3.0e38f; l[i][r] = 0.f; }

  // scores run natively in exp2 domain: kscl = (1/sqrt(64)) * log2(e)
  const float kscl = 0.125f * 1.4426950408889634f;

  issueK(0);                                  // prologue: tile 0 in flight

  for (int kb = 0; kb < NKB; ++kb) {
    const int key0 = kb * 32;
    unsigned short (*kbuf)[80] = kT[kb & 1];
    unsigned short (*vbuf)[48] = vT[kb & 1];

    // ---- V tile: load + transpose into vbuf (overlaps in-flight K DMA).
    for (int i = threadIdx.x; i < 256; i += 128) {
      const int kr = i >> 3, seg = i & 7;
      const unsigned short* src =
          qkv + (rowBase + key0 + kr) * LDQ + h * 192 + 128 + seg * 8;
      uint4 vv = *(const uint4*)src;
      const unsigned short* vs = (const unsigned short*)&vv;
#pragma unroll
      for (int j = 0; j < 8; ++j) vbuf[seg * 8 + j][kr] = vs[j];
    }

    // ---- Pipeline: issue next K tile, then wait only for the current one.
    // ASYNCcnt completes in order; <=2 outstanding => current tile landed.
    if (kb + 1 < NKB) {
      issueK(kb + 1);
      asm volatile("s_wait_asynccnt 0x2" ::: "memory");
    } else {
      asm volatile("s_wait_asynccnt 0x0" ::: "memory");
    }
    __syncthreads();

    // ---- Scores: s[strip][keysub], K-fragments shared across both strips.
    v8f s[2][2];
#pragma unroll
    for (int i = 0; i < 2; ++i)
#pragma unroll
      for (int t = 0; t < 2; ++t)
        s[i][t] = (v8f){0.f,0.f,0.f,0.f,0.f,0.f,0.f,0.f};
#pragma unroll
    for (int c = 0; c < 2; ++c) {
      BF16Frag b0, b1;
      const unsigned short* k0p = &kbuf[lc][c * 32 + lh * 16];
      b0.q[0] = *(const uint4*)k0p;
      b0.q[1] = *(const uint4*)(k0p + 8);
      const unsigned short* k1p = &kbuf[16 + lc][c * 32 + lh * 16];
      b1.q[0] = *(const uint4*)k1p;
      b1.q[1] = *(const uint4*)(k1p + 8);
#pragma unroll
      for (int i = 0; i < 2; ++i) {
        s[i][0] = wmma_bf16(aq[i][c].v, b0.v, s[i][0]);
        s[i][1] = wmma_bf16(aq[i][c].v, b1.v, s[i][1]);
      }
    }

    // ---- Online softmax per strip (DPP reductions within 16-lane halves).
#pragma unroll
    for (int i = 0; i < 2; ++i) {
      float alpha[8];
#pragma unroll
      for (int r = 0; r < 8; ++r) {
        float x0 = s[i][0][r] * kscl;
        float x1 = s[i][1][r] * kscl;
        float mx = dpp_max16(fmaxf(x0, x1));
        float mn = fmaxf(m[i][r], mx);
        alpha[r] = exp2f(m[i][r] - mn);
        m[i][r] = mn;
        float p0 = exp2f(x0 - mn);
        float p1 = exp2f(x1 - mn);
        s[i][0][r] = p0;
        s[i][1][r] = p1;
        float rs = dpp_sum16(p0 + p1);
        l[i][r] = l[i][r] * alpha[r] + rs;
      }
#pragma unroll
      for (int nt = 0; nt < 4; ++nt)
#pragma unroll
        for (int r = 0; r < 8; ++r) accO[i][nt][r] *= alpha[r];
      // P strip -> per-wave LDS (C-fragment layout -> row-major bf16)
#pragma unroll
      for (int r = 0; r < 8; ++r) {
        pB[wid][i * 16 + lh * 8 + r][lc]      = f2bf(s[i][0][r]);
        pB[wid][i * 16 + lh * 8 + r][16 + lc] = f2bf(s[i][1][r]);
      }
    }

    // ---- O += P x V (V-fragments shared across strips)
    BF16Frag ap[2];
#pragma unroll
    for (int i = 0; i < 2; ++i) {
      const unsigned short* pr = &pB[wid][i * 16 + lc][0];
      ap[i].q[0] = *(const uint4*)(pr + lh * 16);
      ap[i].q[1] = *(const uint4*)(pr + lh * 16 + 8);
    }
#pragma unroll
    for (int nt = 0; nt < 4; ++nt) {
      BF16Frag bv;
      const unsigned short* vr = &vbuf[nt * 16 + lc][lh * 16];
      bv.q[0] = *(const uint4*)vr;
      bv.q[1] = *(const uint4*)(vr + 8);
#pragma unroll
      for (int i = 0; i < 2; ++i)
        accO[i][nt] = wmma_bf16(ap[i].v, bv.v, accO[i][nt]);
    }
    __syncthreads();
  }

  // ---- Normalize and write concat-head output.
#pragma unroll
  for (int i = 0; i < 2; ++i) {
    float inv[8];
#pragma unroll
    for (int r = 0; r < 8; ++r) inv[r] = 1.0f / l[i][r];
#pragma unroll
    for (int nt = 0; nt < 4; ++nt) {
      const int col = h * 64 + nt * 16 + lc;
#pragma unroll
      for (int r = 0; r < 8; ++r) {
        const int qrow = q0 + i * 16 + lh * 8 + r;
        out[(rowBase + qrow) * FEA + col] = f2bf(accO[i][nt][r] * inv[r]);
      }
    }
  }
}

// ---------------------------------------------------------------------------
extern "C" void kernel_launch(void* const* d_in, const int* in_sizes, int n_in,
                              void* d_out, int out_size, void* d_ws, size_t ws_size,
                              hipStream_t stream) {
  const float* x     = (const float*)d_in[0];   // [2,2048,1024]
  const float* qkv_w = (const float*)d_in[1];   // [3072,1024]
  const float* out_w = (const float*)d_in[2];   // [1024,1024]
  const float* out_b = (const float*)d_in[3];   // [1024]
  float* out = (float*)d_out;                   // [2,2048,1024] f32

  constexpr int M   = 2 * 2048;   // 4096
  constexpr int D   = 1024;
  constexpr int N1  = 3072;
  constexpr int FEA = 1024;

  unsigned short* xb    = (unsigned short*)d_ws;              // M*D
  unsigned short* wqkvb = xb + (size_t)M * D;                 // N1*D
  unsigned short* wob   = wqkvb + (size_t)N1 * D;             // D*FEA
  unsigned short* qkvb  = wob + (size_t)D * FEA;              // M*N1
  unsigned short* attnb = qkvb + (size_t)M * N1;              // M*FEA

  cvt_f32_to_bf16<<<(M * D / 4 + 255) / 256, 256, 0, stream>>>(x, xb, M * D / 4);
  cvt_f32_to_bf16<<<(N1 * D / 4 + 255) / 256, 256, 0, stream>>>(qkv_w, wqkvb, N1 * D / 4);
  cvt_f32_to_bf16<<<(D * FEA / 4 + 255) / 256, 256, 0, stream>>>(out_w, wob, D * FEA / 4);

  gemm_bf16_nt<false><<<dim3(N1 / 256, M / 64), 256, 0, stream>>>(
      xb, wqkvb, qkvb, nullptr, M, N1, D);

  attn_flash<<<dim3(16, 16, 2), 128, 0, stream>>>(qkvb, attnb);

  gemm_bf16_nt<true><<<dim3(FEA / 256, M / 64), 256, 0, stream>>>(
      attnb, wob, out, out_b, M, FEA, D);
}
